// FinH2AN_31653908971672
// MI455X (gfx1250) — matrix operand order repaired and checked by reading, not compiled, not run on hardware
//
#include <hip/hip_runtime.h>

typedef __attribute__((ext_vector_type(16))) _Float16 v16h;
typedef __attribute__((ext_vector_type(8)))  float    v8f;
typedef _Float16 f16;

#define MACCS 167
#define PUBCH 881
#define ERGN  441
#define FPAD  1536   // 1489 padded to mult of 64 (even number of K=32 steps)
#define HE    512
#define EDIM  128
#define QDIM  64
#define PROJ  256
#define HIDR  300
#define HIDP  320    // 300 padded

union FragU { v16h v; uint4 q[2]; };
struct Frags { FragU a0, a1, b0, b1, b2, b3; };   // named members only -> stays in VGPRs

enum { EPI_BIAS = 0, EPI_BIAS_RELU = 1, EPI_SCORE = 2, EPI_BIAS_ADD = 3 };

// Generic f16 WMMA GEMM: C[M,N] = epilogue(A[M,K] @ BT[N,K]^T)
// 256 threads = 8 waves; block tile 256(M) x 64(N); wave tile 32 x 64.
// Ping/pong register pipeline over K (step 32, unrolled x2; K % 64 == 0).
template<int EPI, typename OutT>
__global__ __launch_bounds__(256)
void gemm_wmma_kernel(const f16* __restrict__ A, int lda,
                      const f16* __restrict__ BT, int ldb,
                      const float* __restrict__ bias, int nbias,
                      OutT* __restrict__ C, int ldc,
                      const f16* __restrict__ addend, int ldadd,
                      const int* __restrict__ mask, int mask_ld, int mask_n,
                      int M, int N, int K)
{
    const int lane = threadIdx.x & 31;
    const int wave = threadIdx.x >> 5;
    const int rowBase = blockIdx.y * 256 + wave * 32;
    const int colBase = blockIdx.x * 64;
    if (rowBase >= M) return;

    const int mr = lane & 15;  // matrix row (A) / col (B) within 16-tile
    const int g  = lane >> 4;  // K-half group

    // Per-lane base pointers (fragment layouts per gfx1250 ISA):
    // A 16x32: lane=row mr, group g -> two b128 chunks at k+8g and k+16+8g
    const f16* ap0 = A + (size_t)(rowBase + mr) * lda + 8 * g;
    const f16* ap1 = A + (size_t)(rowBase + 16 + mr) * lda + 8 * g;
    // B 32x16: lane=col, group g -> two b128 chunks at k+8g and k+16+8g
    const f16* bp0 = BT + (size_t)(colBase +  0 + mr) * ldb + 8 * g;
    const f16* bp1 = BT + (size_t)(colBase + 16 + mr) * ldb + 8 * g;
    const f16* bp2 = BT + (size_t)(colBase + 32 + mr) * ldb + 8 * g;
    const f16* bp3 = BT + (size_t)(colBase + 48 + mr) * ldb + 8 * g;

#define LOAD_FRAGS(F, KB) do {                                          \
        F.a0.q[0] = *(const uint4*)(ap0 + (KB));                        \
        F.a0.q[1] = *(const uint4*)(ap0 + (KB) + 16);                   \
        F.a1.q[0] = *(const uint4*)(ap1 + (KB));                        \
        F.a1.q[1] = *(const uint4*)(ap1 + (KB) + 16);                   \
        F.b0.q[0] = *(const uint4*)(bp0 + (KB));                        \
        F.b0.q[1] = *(const uint4*)(bp0 + (KB) + 16);                   \
        F.b1.q[0] = *(const uint4*)(bp1 + (KB));                        \
        F.b1.q[1] = *(const uint4*)(bp1 + (KB) + 16);                   \
        F.b2.q[0] = *(const uint4*)(bp2 + (KB));                        \
        F.b2.q[1] = *(const uint4*)(bp2 + (KB) + 16);                   \
        F.b3.q[0] = *(const uint4*)(bp3 + (KB));                        \
        F.b3.q[1] = *(const uint4*)(bp3 + (KB) + 16);                   \
    } while (0)

#define WMMA_FRAGS(F) do {                                              \
        acc[0] = __builtin_amdgcn_wmma_f32_16x16x32_f16(                \
            false, F.a0.v, false, F.b0.v, (short)0, acc[0], false, false); \
        acc[4] = __builtin_amdgcn_wmma_f32_16x16x32_f16(                \
            false, F.a1.v, false, F.b0.v, (short)0, acc[4], false, false); \
        acc[1] = __builtin_amdgcn_wmma_f32_16x16x32_f16(                \
            false, F.a0.v, false, F.b1.v, (short)0, acc[1], false, false); \
        acc[5] = __builtin_amdgcn_wmma_f32_16x16x32_f16(                \
            false, F.a1.v, false, F.b1.v, (short)0, acc[5], false, false); \
        acc[2] = __builtin_amdgcn_wmma_f32_16x16x32_f16(                \
            false, F.a0.v, false, F.b2.v, (short)0, acc[2], false, false); \
        acc[6] = __builtin_amdgcn_wmma_f32_16x16x32_f16(                \
            false, F.a1.v, false, F.b2.v, (short)0, acc[6], false, false); \
        acc[3] = __builtin_amdgcn_wmma_f32_16x16x32_f16(                \
            false, F.a0.v, false, F.b3.v, (short)0, acc[3], false, false); \
        acc[7] = __builtin_amdgcn_wmma_f32_16x16x32_f16(                \
            false, F.a1.v, false, F.b3.v, (short)0, acc[7], false, false); \
    } while (0)

    v8f acc[8] = {};
    Frags fa, fb;
    LOAD_FRAGS(fa, 0);
    for (int kb = 0; kb < K; kb += 64) {
        LOAD_FRAGS(fb, kb + 32);          // always valid: K % 64 == 0
        WMMA_FRAGS(fa);
        if (kb + 64 < K) LOAD_FRAGS(fa, kb + 64);
        WMMA_FRAGS(fb);
    }
#undef LOAD_FRAGS
#undef WMMA_FRAGS

    // D layout: lane -> col (lane&15); VGPR r -> row 8*(lane>>4)+r
    #pragma unroll
    for (int half = 0; half < 2; ++half) {
        #pragma unroll
        for (int t = 0; t < 4; ++t) {
            const int ncol = colBase + t * 16 + mr;
            const float bv = (ncol < nbias) ? bias[ncol] : 0.0f;
            #pragma unroll
            for (int r = 0; r < 8; ++r) {
                const int mrow = rowBase + half * 16 + 8 * g + r;
                float v = acc[half * 4 + t][r];
                if (EPI == EPI_SCORE) {
                    v *= 0.125f;                      // 1/sqrt(Q)
                    v = (v > 0.0f) ? v : 0.01f * v;   // leaky_relu
                    const bool edge = (ncol < mask_n) &&
                                      (mask[(size_t)mrow * mask_ld + ncol] != 0);
                    if (!edge) v = -1e9f;
                } else {
                    v += bv;
                    if (EPI == EPI_BIAS_RELU) v = fmaxf(v, 0.0f);
                    if (EPI == EPI_BIAS_ADD)
                        v += (float)addend[(size_t)mrow * ldadd + ncol];
                }
                C[(size_t)mrow * ldc + ncol] = (OutT)v;
            }
        }
    }
}

// Pack the three int fingerprint arrays into one padded f16 matrix [B, FPAD]
__global__ void build_x_kernel(const int* __restrict__ ma, const int* __restrict__ pb,
                               const int* __restrict__ er, f16* __restrict__ X, int total)
{
    int idx = blockIdx.x * blockDim.x + threadIdx.x;
    if (idx >= total) return;
    int b = idx / FPAD, k = idx - b * FPAD;
    int v = 0;
    if (k < MACCS)                     v = ma[b * MACCS + k];
    else if (k < MACCS + PUBCH)        v = pb[b * PUBCH + (k - MACCS)];
    else if (k < MACCS + PUBCH + ERGN) v = er[b * ERGN + (k - MACCS - PUBCH)];
    X[idx] = (f16)(float)v;
}

// W[K,N] f32 row-major -> WT[Npad, Kpad] f16 (zero padded)
__global__ void wtrans_kernel(const float* __restrict__ W, f16* __restrict__ WT,
                              int K, int N, int Kpad, int total)
{
    int idx = blockIdx.x * blockDim.x + threadIdx.x;
    if (idx >= total) return;
    int n = idx / Kpad, k = idx - n * Kpad;
    float v = (n < N && k < K) ? W[(size_t)k * N + n] : 0.0f;
    WT[idx] = (f16)v;
}

// k2 = (W5+b5) @ w6 + b6 -> f16 [nbitsPad, 128] (zero rows beyond nbits)
__global__ void make_k2_kernel(const float* __restrict__ W5, const float* __restrict__ b5,
                               const float* __restrict__ W6, const float* __restrict__ b6,
                               f16* __restrict__ K2, int nbits, int total)
{
    int idx = blockIdx.x * blockDim.x + threadIdx.x;
    if (idx >= total) return;
    int i = idx / EDIM, j = idx - i * EDIM;
    float s = 0.0f;
    if (i < nbits) {
        s = b6[j];
        for (int q = 0; q < QDIM; ++q)
            s += (W5[i * QDIM + q] + b5[q]) * W6[q * EDIM + j];
    }
    K2[idx] = (f16)s;
}

// v2^T = ((W5+b5) @ w7 + b7)^T -> f16 [128, nbitsPad]
__global__ void make_v2t_kernel(const float* __restrict__ W5, const float* __restrict__ b5,
                                const float* __restrict__ W7, const float* __restrict__ b7,
                                f16* __restrict__ V2T, int nbits, int nbitsPad, int total)
{
    int idx = blockIdx.x * blockDim.x + threadIdx.x;
    if (idx >= total) return;
    int j = idx / nbitsPad, i = idx - j * nbitsPad;
    float s = 0.0f;
    if (i < nbits) {
        s = b7[j];
        for (int q = 0; q < QDIM; ++q)
            s += (W5[i * QDIM + q] + b5[q]) * W7[q * EDIM + j];
    }
    V2T[idx] = (f16)s;
}

// Masked-dense softmax, one wave32 per row; scores already hold -1e9 for non-edges
__global__ __launch_bounds__(256)
void softmax_kernel(const float* __restrict__ S, f16* __restrict__ P, int ncols, int nrows)
{
    int row = blockIdx.x * 8 + (threadIdx.x >> 5);
    int lane = threadIdx.x & 31;
    if (row >= nrows) return;
    const float* s = S + (size_t)row * ncols;
    float mx = -3.4e38f;
    for (int c = lane; c < ncols; c += 32) mx = fmaxf(mx, s[c]);
    for (int o = 16; o > 0; o >>= 1) mx = fmaxf(mx, __shfl_xor(mx, o, 32));
    float sum = 0.0f;
    for (int c = lane; c < ncols; c += 32) sum += __expf(s[c] - mx);
    for (int o = 16; o > 0; o >>= 1) sum += __shfl_xor(sum, o, 32);
    float inv = 1.0f / sum;
    f16* p = P + (size_t)row * ncols;
    for (int c = lane; c < ncols; c += 32) p[c] = (f16)(__expf(s[c] - mx) * inv);
}

// out[b] = sigmoid(a3[b,0:300] . ffn2W + ffn2b); one wave per row
__global__ __launch_bounds__(256)
void final_kernel(const f16* __restrict__ A3, const float* __restrict__ W,
                  const float* __restrict__ b, float* __restrict__ out, int nrows)
{
    int row = blockIdx.x * 8 + (threadIdx.x >> 5);
    int lane = threadIdx.x & 31;
    if (row >= nrows) return;
    const f16* a = A3 + (size_t)row * HIDP;
    float s = 0.0f;
    for (int k = lane; k < HIDR; k += 32) s += (float)a[k] * W[k];
    for (int o = 16; o > 0; o >>= 1) s += __shfl_xor(s, o, 32);
    if (lane == 0) out[row] = 1.0f / (1.0f + __expf(-(s + b[0])));
}

static inline size_t alignUp256(size_t x) { return (x + 255) & ~(size_t)255; }

extern "C" void kernel_launch(void* const* d_in, const int* in_sizes, int n_in,
                              void* d_out, int out_size, void* d_ws, size_t ws_size,
                              hipStream_t stream)
{
    const int* maccs = (const int*)d_in[0];
    const int* pubch = (const int*)d_in[1];
    const int* ergfp = (const int*)d_in[2];
    const int B = in_sizes[0] / MACCS;

    const float* w1W = (const float*)d_in[3];  const float* w1b = (const float*)d_in[4];
    const float* w2W = (const float*)d_in[5];  const float* w2b = (const float*)d_in[6];
    // d_in[7..10] = w3, w4 (dead branch: skipped)
    const float* w5W[3] = {(const float*)d_in[11], (const float*)d_in[13], (const float*)d_in[15]};
    const float* w5b[3] = {(const float*)d_in[12], (const float*)d_in[14], (const float*)d_in[16]};
    const float* w6W[3] = {(const float*)d_in[17], (const float*)d_in[21], (const float*)d_in[25]};
    const float* w6b[3] = {(const float*)d_in[18], (const float*)d_in[22], (const float*)d_in[26]};
    const float* w7W[3] = {(const float*)d_in[19], (const float*)d_in[23], (const float*)d_in[27]};
    const float* w7b[3] = {(const float*)d_in[20], (const float*)d_in[24], (const float*)d_in[28]};
    const float* mlp1W = (const float*)d_in[29]; const float* mlp1b = (const float*)d_in[30];
    const float* mlp2W = (const float*)d_in[31]; const float* mlp2b = (const float*)d_in[32];
    const float* resW  = (const float*)d_in[33]; const float* resb  = (const float*)d_in[34];
    const float* fc1W  = (const float*)d_in[35]; const float* fc1b  = (const float*)d_in[36];
    const float* fc2W  = (const float*)d_in[37]; const float* fc2b  = (const float*)d_in[38];
    const float* ffn1W = (const float*)d_in[39]; const float* ffn1b = (const float*)d_in[40];
    const float* ffn2W = (const float*)d_in[41]; const float* ffn2b = (const float*)d_in[42];

    const int nbits[3] = {MACCS, PUBCH, ERGN};
    const int nbpad[3] = {192, 896, 448};          // padded to mult of 64
    const int* fpPtr[3] = {maccs, pubch, ergfp};

    // ---- workspace carve ----
    char* ws = (char*)d_ws;
    size_t off = 0;
    auto alloc = [&](size_t bytes) { void* p = ws + off; off = alignUp256(off + bytes); return p; };
    f16* X     = (f16*)alloc((size_t)B * FPAD * 2);
    f16* w1T   = (f16*)alloc((size_t)HE * FPAD * 2);
    f16* efeat = (f16*)alloc((size_t)B * HE * 2);
    f16* w2T   = (f16*)alloc((size_t)EDIM * HE * 2);
    f16* qfrom = (f16*)alloc((size_t)B * EDIM * 2);
    f16* k2[3]; f16* v2t[3];
    for (int t = 0; t < 3; ++t) {
        k2[t]  = (f16*)alloc((size_t)nbpad[t] * EDIM * 2);
        v2t[t] = (f16*)alloc((size_t)EDIM * nbpad[t] * 2);
    }
    float* scores = (float*)alloc((size_t)B * 896 * 4);   // reused per type
    f16*   probs  = (f16*)alloc((size_t)B * 896 * 2);     // reused per type
    f16* henew    = (f16*)alloc((size_t)B * 384 * 2);
    f16* mlp1T    = (f16*)alloc((size_t)PROJ * 384 * 2);
    f16* mlp2T    = (f16*)alloc((size_t)EDIM * PROJ * 2);
    f16* h1       = (f16*)alloc((size_t)B * PROJ * 2);
    f16* fpemb    = (f16*)alloc((size_t)B * EDIM * 2);
    f16* resT     = (f16*)alloc((size_t)EDIM * FPAD * 2);
    f16* fused    = (f16*)alloc((size_t)B * EDIM * 2);
    f16* fc1T     = (f16*)alloc((size_t)PROJ * EDIM * 2);
    f16* a1       = (f16*)alloc((size_t)B * PROJ * 2);
    f16* fc2T     = (f16*)alloc((size_t)HIDP * PROJ * 2);
    f16* a2       = (f16*)alloc((size_t)B * HIDP * 2);
    f16* ffn1T    = (f16*)alloc((size_t)HIDP * HIDP * 2);
    f16* a3       = (f16*)alloc((size_t)B * HIDP * 2);
    (void)ws_size; (void)n_in; (void)out_size;

    const int TB = 256;
    auto blks = [](int total) { return (total + 255) / 256; };
    auto g2 = [](int N, int M) { return dim3((unsigned)(N / 64), (unsigned)((M + 255) / 256)); };

    // ---- preprocessing ----
    { int t = B * FPAD; build_x_kernel<<<blks(t), TB, 0, stream>>>(maccs, pubch, ergfp, X, t); }
    { int t = HE * FPAD;   wtrans_kernel<<<blks(t), TB, 0, stream>>>(w1W,  w1T,  1489, HE,   FPAD, t); }
    { int t = EDIM * HE;   wtrans_kernel<<<blks(t), TB, 0, stream>>>(w2W,  w2T,  HE,   EDIM, HE,   t); }
    { int t = PROJ * 384;  wtrans_kernel<<<blks(t), TB, 0, stream>>>(mlp1W, mlp1T, 384, PROJ, 384,  t); }
    { int t = EDIM * PROJ; wtrans_kernel<<<blks(t), TB, 0, stream>>>(mlp2W, mlp2T, PROJ, EDIM, PROJ, t); }
    { int t = EDIM * FPAD; wtrans_kernel<<<blks(t), TB, 0, stream>>>(resW, resT, 1489, EDIM, FPAD, t); }
    { int t = PROJ * EDIM; wtrans_kernel<<<blks(t), TB, 0, stream>>>(fc1W, fc1T, EDIM, PROJ, EDIM, t); }
    { int t = HIDP * PROJ; wtrans_kernel<<<blks(t), TB, 0, stream>>>(fc2W, fc2T, PROJ, HIDR, PROJ, t); }
    { int t = HIDP * HIDP; wtrans_kernel<<<blks(t), TB, 0, stream>>>(ffn1W, ffn1T, HIDR, HIDR, HIDP, t); }
    for (int t = 0; t < 3; ++t) {
        int tk = nbpad[t] * EDIM;
        make_k2_kernel<<<blks(tk), TB, 0, stream>>>(w5W[t], w5b[t], w6W[t], w6b[t], k2[t], nbits[t], tk);
        make_v2t_kernel<<<blks(tk), TB, 0, stream>>>(w5W[t], w5b[t], w7W[t], w7b[t], v2t[t], nbits[t], nbpad[t], tk);
    }

    // ---- main WMMA chain ----
    gemm_wmma_kernel<EPI_BIAS, f16><<<g2(HE, B), TB, 0, stream>>>(
        X, FPAD, w1T, FPAD, w1b, HE, efeat, HE,
        nullptr, 0, nullptr, 0, 0, B, HE, FPAD);

    gemm_wmma_kernel<EPI_BIAS, f16><<<g2(EDIM, B), TB, 0, stream>>>(
        efeat, HE, w2T, HE, w2b, EDIM, qfrom, EDIM,
        nullptr, 0, nullptr, 0, 0, B, EDIM, HE);

    for (int t = 0; t < 3; ++t) {
        const int np = nbpad[t], nb = nbits[t];
        gemm_wmma_kernel<EPI_SCORE, float><<<g2(np, B), TB, 0, stream>>>(
            qfrom, EDIM, k2[t], EDIM, nullptr, 0, scores, np,
            nullptr, 0, fpPtr[t], nb, nb, B, np, EDIM);
        softmax_kernel<<<(B + 7) / 8, TB, 0, stream>>>(scores, probs, np, B);
        gemm_wmma_kernel<EPI_BIAS, f16><<<g2(EDIM, B), TB, 0, stream>>>(
            probs, np, v2t[t], np, nullptr, 0, henew + t * EDIM, 384,
            nullptr, 0, nullptr, 0, 0, B, EDIM, np);
    }

    gemm_wmma_kernel<EPI_BIAS_RELU, f16><<<g2(PROJ, B), TB, 0, stream>>>(
        henew, 384, mlp1T, 384, mlp1b, PROJ, h1, PROJ,
        nullptr, 0, nullptr, 0, 0, B, PROJ, 384);

    gemm_wmma_kernel<EPI_BIAS_RELU, f16><<<g2(EDIM, B), TB, 0, stream>>>(
        h1, PROJ, mlp2T, PROJ, mlp2b, EDIM, fpemb, EDIM,
        nullptr, 0, nullptr, 0, 0, B, EDIM, PROJ);

    gemm_wmma_kernel<EPI_BIAS_ADD, f16><<<g2(EDIM, B), TB, 0, stream>>>(
        X, FPAD, resT, FPAD, resb, EDIM, fused, EDIM,
        fpemb, EDIM, nullptr, 0, 0, B, EDIM, FPAD);

    gemm_wmma_kernel<EPI_BIAS_RELU, f16><<<g2(PROJ, B), TB, 0, stream>>>(
        fused, EDIM, fc1T, EDIM, fc1b, PROJ, a1, PROJ,
        nullptr, 0, nullptr, 0, 0, B, PROJ, EDIM);

    gemm_wmma_kernel<EPI_BIAS, f16><<<g2(HIDP, B), TB, 0, stream>>>(
        a1, PROJ, fc2T, PROJ, fc2b, HIDR, a2, HIDP,
        nullptr, 0, nullptr, 0, 0, B, HIDP, PROJ);

    gemm_wmma_kernel<EPI_BIAS_RELU, f16><<<g2(HIDP, B), TB, 0, stream>>>(
        a2, HIDP, ffn1T, HIDP, ffn1b, HIDR, a3, HIDP,
        nullptr, 0, nullptr, 0, 0, B, HIDP, HIDP);

    final_kernel<<<(B + 7) / 8, TB, 0, stream>>>(a3, ffn2W, ffn2b, (float*)d_out, B);
}
